// CausalSelfAttentionTF_16028817949063
// MI455X (gfx1250) — compile-verified, matrix operand order checked by reading, not run on hardware
//
#include <hip/hip_runtime.h>

// ---------------------------------------------------------------------------
// Causal self-attention (GPT-style) for MI455X / gfx1250.
//   B=2, T=2048, D=1024, H=16, hd=64
// Stage 1: QKV = x @ w_attn + b_attn        (bf16 WMMA, f32 accum, pipelined)
// Stage 2: flash attention per (b,h)        (bf16 WMMA, online softmax)
// Stage 3: out = y @ w_proj + b_proj        (bf16 WMMA, f32 accum, pipelined)
// ---------------------------------------------------------------------------

typedef __bf16 bf16;
typedef __attribute__((ext_vector_type(16))) __bf16 v16bf;
typedef __attribute__((ext_vector_type(8)))  __bf16 v8bf;
typedef __attribute__((ext_vector_type(4)))  __bf16 v4bf;
typedef __attribute__((ext_vector_type(8)))  float  v8f;

#define B_     2
#define T_     2048
#define H_     16
#define HD_    64
#define D_     1024
#define N3_    3072
#define M_     4096   // B*T

// Native f32 -> bf16 convert (gfx1250 has hardware bf16 converts).
__device__ __forceinline__ bf16 cvt_bf16(float f) { return (bf16)f; }

// Load one 16x32 bf16 fragment (A-style; B mirrored with lane = N column).
// Per CDNA5 ISA 7.12.2: lane L (0-15) holds row M=L, K = {hl*8..+7, 16+hl*8..+7}
// as two contiguous 16-byte chunks -> two ds_load_b128 / global_load_b128.
__device__ __forceinline__ v16bf load_frag(const bf16* base, int ld, int lane) {
  const int row = lane & 15;
  const int hl  = (lane >> 4) & 1;
  const bf16* p = base + row * ld + hl * 8;
  v8bf lo = *(const v8bf*)(p);
  v8bf hi = *(const v8bf*)(p + 16);
  return __builtin_shufflevector(lo, hi, 0, 1, 2, 3, 4, 5, 6, 7,
                                 8, 9, 10, 11, 12, 13, 14, 15);
}

__device__ __forceinline__ v8f wmma_bf16(v16bf a, v16bf b, v8f c) {
  return __builtin_amdgcn_wmma_f32_16x16x32_bf16(
      /*neg_a=*/false, a, /*neg_b=*/false, b,
      /*c_mod=*/(short)0, c, /*reuse_a=*/false, /*reuse_b=*/false);
}

// ---------------------------------------------------------------------------
// Shared GEMM mainloop body (register-pipelined, 128x128x32 block tiles,
// 8 waves, wave tile 32x64 = 2x4 WMMA accumulators).
// ---------------------------------------------------------------------------
#define GEMM_PROLOGUE(Nc)                                                     \
  const int tid  = threadIdx.x;                                               \
  const int lane = tid & 31;                                                  \
  const int wave = tid >> 5;                                                  \
  const int wm   = wave >> 1;                                                 \
  const int wn   = wave & 1;                                                  \
  const int m0   = blockIdx.y * 128;                                          \
  const int n0   = blockIdx.x * 128;                                          \
  v8f acc[2][4];                                                              \
  _Pragma("unroll") for (int i = 0; i < 2; i++)                               \
  _Pragma("unroll") for (int j = 0; j < 4; j++) acc[i][j] = {};               \
  float4 ra[4], rb[4];                                                        \
  /* prologue: fetch k0 = 0 tiles into registers */                           \
  _Pragma("unroll") for (int i = 0; i < 4; i++) {                             \
    int e = tid + i * 256;                                                    \
    ra[i] = *(const float4*)(A + (size_t)(m0 + (e >> 3)) * D_ + (e & 7) * 4); \
    rb[i] = *(const float4*)(Bw + (size_t)(e >> 5) * (Nc) + n0 + (e & 31) * 4);\
  }                                                                           \
  for (int k0 = 0; k0 < D_; k0 += 32) {                                       \
    /* stage registers -> LDS (fp32 -> bf16) */                               \
    _Pragma("unroll") for (int i = 0; i < 4; i++) {                           \
      int e = tid + i * 256;                                                  \
      v4bf p;                                                                 \
      p[0] = cvt_bf16(ra[i].x); p[1] = cvt_bf16(ra[i].y);                     \
      p[2] = cvt_bf16(ra[i].z); p[3] = cvt_bf16(ra[i].w);                     \
      *(v4bf*)&As[e >> 3][(e & 7) * 4] = p;                                   \
      int kr = e >> 5, c4 = e & 31;                                           \
      Bs[c4 * 4 + 0][kr] = cvt_bf16(rb[i].x);                                 \
      Bs[c4 * 4 + 1][kr] = cvt_bf16(rb[i].y);                                 \
      Bs[c4 * 4 + 2][kr] = cvt_bf16(rb[i].z);                                 \
      Bs[c4 * 4 + 3][kr] = cvt_bf16(rb[i].w);                                 \
    }                                                                         \
    __syncthreads();                                                          \
    if (k0 + 32 < D_) { /* prefetch next K-step while WMMA runs */            \
      _Pragma("unroll") for (int i = 0; i < 4; i++) {                         \
        int e = tid + i * 256;                                                \
        ra[i] = *(const float4*)(A + (size_t)(m0 + (e >> 3)) * D_ +           \
                                 k0 + 32 + (e & 7) * 4);                      \
        rb[i] = *(const float4*)(Bw + (size_t)(k0 + 32 + (e >> 5)) * (Nc) +   \
                                 n0 + (e & 31) * 4);                          \
      }                                                                       \
    }                                                                         \
    v16bf af[2], bfr[4];                                                      \
    _Pragma("unroll") for (int im = 0; im < 2; im++)                          \
      af[im] = load_frag(&As[wm * 32 + im * 16][0], 32, lane);                \
    _Pragma("unroll") for (int jn = 0; jn < 4; jn++)                          \
      bfr[jn] = load_frag(&Bs[wn * 64 + jn * 16][0], 32, lane);               \
    _Pragma("unroll") for (int im = 0; im < 2; im++)                          \
    _Pragma("unroll") for (int jn = 0; jn < 4; jn++)                          \
      acc[im][jn] = wmma_bf16(af[im], bfr[jn], acc[im][jn]);                  \
    __syncthreads();                                                          \
  }

// ---------------------------------------------------------------------------
// Stage 1: QKV GEMM -> scatter into bf16 q/k/v workspaces [B,H,T,hd].
// ---------------------------------------------------------------------------
__global__ __launch_bounds__(256) void qkv_gemm_kernel(
    const float* __restrict__ A, const float* __restrict__ Bw,
    const float* __restrict__ bias,
    bf16* __restrict__ q_ws, bf16* __restrict__ k_ws, bf16* __restrict__ v_ws) {
  __shared__ bf16 As[128][32];
  __shared__ bf16 Bs[128][32];   // N-major (transposed)
  GEMM_PROLOGUE(N3_)

  const int hl = lane >> 4;
  const int nl = lane & 15;
#pragma unroll
  for (int im = 0; im < 2; im++) {
#pragma unroll
    for (int jn = 0; jn < 4; jn++) {
      int gn    = n0 + wn * 64 + jn * 16 + nl;
      int which = gn >> 10;           // 0=q 1=k 2=v (16-wide tile never straddles)
      int rem   = gn & 1023;
      int h     = rem >> 6;
      int d     = rem & 63;
      float bv  = bias[gn];
      bf16* dst = (which == 0) ? q_ws : ((which == 1) ? k_ws : v_ws);
#pragma unroll
      for (int r = 0; r < 8; r++) {
        int gm = m0 + wm * 32 + im * 16 + r + 8 * hl;
        int b  = gm >> 11;            // / T_
        int t  = gm & (T_ - 1);
        size_t off = ((size_t)(b * H_ + h) * T_ + t) * HD_ + d;
        dst[off] = cvt_bf16(acc[im][jn][r] + bv);
      }
    }
  }
}

// ---------------------------------------------------------------------------
// Stage 3: output projection.  out[4096,1024] = y @ w_proj + b_proj (fp32)
// ---------------------------------------------------------------------------
__global__ __launch_bounds__(256) void proj_gemm_kernel(
    const float* __restrict__ A, const float* __restrict__ Bw,
    const float* __restrict__ bias, float* __restrict__ C) {
  __shared__ bf16 As[128][32];
  __shared__ bf16 Bs[128][32];
  GEMM_PROLOGUE(D_)

  const int hl = lane >> 4;
  const int nl = lane & 15;
#pragma unroll
  for (int im = 0; im < 2; im++) {
#pragma unroll
    for (int jn = 0; jn < 4; jn++) {
      int gn   = n0 + wn * 64 + jn * 16 + nl;
      float bv = bias[gn];
#pragma unroll
      for (int r = 0; r < 8; r++) {
        int gm = m0 + wm * 32 + im * 16 + r + 8 * hl;
        C[(size_t)gm * D_ + gn] = acc[im][jn][r] + bv;
      }
    }
  }
}

// ---------------------------------------------------------------------------
// Stage 2: flash attention.  grid = (T/128, B*H), 8 waves per block,
// each wave owns 16 query rows; online softmax over 64-key tiles.
// ---------------------------------------------------------------------------
__global__ __launch_bounds__(256) void attn_kernel(
    const bf16* __restrict__ q_ws, const bf16* __restrict__ k_ws,
    const bf16* __restrict__ v_ws, float* __restrict__ y) {
  __shared__ bf16 Ks[64][64];        // [key][hd]
  __shared__ bf16 Vt[64][64];        // [hd][key]  (transposed V)
  __shared__ bf16 Ps[8][16][64];     // per-wave P relayout buffer

  const int tid   = threadIdx.x;
  const int lane  = tid & 31;
  const int wave  = tid >> 5;                    // 0..7
  const int hl    = lane >> 4;
  const int nl    = lane & 15;
  const int qtile = blockIdx.x;                  // 0..15 (128 q-rows each)
  const int bh    = blockIdx.y;                  // b*H + h
  const int qbase = qtile * 128;
  const int qw    = qbase + wave * 16;           // wave's first q row
  const size_t base = (size_t)bh * T_ * HD_;
  const float scale = 0.125f;                    // 1/sqrt(64)

  // Q fragments: rows qw + (lane&15); hd chunks {0,32}
  v16bf qf[2];
  {
    const bf16* qp = q_ws + base + (size_t)qw * HD_;
    qf[0] = load_frag(qp + 0,  HD_, lane);
    qf[1] = load_frag(qp + 32, HD_, lane);
  }

  v8f o[4];
#pragma unroll
  for (int dn = 0; dn < 4; dn++) o[dn] = {};
  float mstate[8], lstate[8];
#pragma unroll
  for (int r = 0; r < 8; r++) { mstate[r] = -1e30f; lstate[r] = 0.0f; }

  const int jtmax = 2 * qtile + 1;               // keys up to qbase+127
  for (int jt = 0; jt <= jtmax; jt++) {          // uniform per block
    const int k0 = jt * 64;
    __syncthreads();                             // prev iter's LDS reads done
    // --- K tile: contiguous 8KB, flat copy ---
    {
      const uint4* src = (const uint4*)(k_ws + base + (size_t)k0 * HD_);
      uint4* dst = (uint4*)&Ks[0][0];
#pragma unroll
      for (int i = 0; i < 2; i++) dst[tid + i * 256] = src[tid + i * 256];
    }
    // --- V tile: transpose into Vt[hd][key] ---
    {
      const bf16* src = v_ws + base + (size_t)k0 * HD_;
      int key = tid >> 2;                        // 0..63
      int d0  = (tid & 3) * 16;
#pragma unroll
      for (int c = 0; c < 2; c++) {
        v8bf t8 = *(const v8bf*)(src + key * 64 + d0 + c * 8);
#pragma unroll
        for (int e = 0; e < 8; e++) Vt[d0 + c * 8 + e][key] = t8[e];
      }
    }
    __syncthreads();

    // --- S = Q @ K^T : 16 q-rows x 64 keys ---
    v8f s[4];
#pragma unroll
    for (int jn = 0; jn < 4; jn++) s[jn] = {};
#pragma unroll
    for (int kk = 0; kk < 2; kk++) {
#pragma unroll
      for (int jn = 0; jn < 4; jn++) {
        v16bf kf = load_frag(&Ks[jn * 16][kk * 32], 64, lane);
        s[jn] = wmma_bf16(qf[kk], kf, s[jn]);
      }
    }

    // --- scale + causal mask (wave-uniform predicate; EXEC stays full) ---
    const bool needmask = (k0 + 63) > qw;
#pragma unroll
    for (int jn = 0; jn < 4; jn++) {
#pragma unroll
      for (int r = 0; r < 8; r++) {
        float v = s[jn][r] * scale;
        if (needmask) {
          int kg = k0 + jn * 16 + nl;
          int qg = qw + r + 8 * hl;
          if (kg > qg) v = -1e30f;
        }
        s[jn][r] = v;
      }
    }

    // --- online softmax (row = r + 8*hl, cols striped across 16 lanes) ---
    float alpha[8];
#pragma unroll
    for (int r = 0; r < 8; r++) {
      float tm = s[0][r];
#pragma unroll
      for (int jn = 1; jn < 4; jn++) tm = fmaxf(tm, s[jn][r]);
      tm = fmaxf(tm, __shfl_xor(tm, 1, 32));
      tm = fmaxf(tm, __shfl_xor(tm, 2, 32));
      tm = fmaxf(tm, __shfl_xor(tm, 4, 32));
      tm = fmaxf(tm, __shfl_xor(tm, 8, 32));
      float mn = fmaxf(mstate[r], tm);
      alpha[r] = __expf(mstate[r] - mn);
      mstate[r] = mn;
      float rs = 0.0f;
#pragma unroll
      for (int jn = 0; jn < 4; jn++) {
        float p = __expf(s[jn][r] - mn);
        s[jn][r] = p;
        rs += p;
      }
      rs += __shfl_xor(rs, 1, 32);
      rs += __shfl_xor(rs, 2, 32);
      rs += __shfl_xor(rs, 4, 32);
      rs += __shfl_xor(rs, 8, 32);
      lstate[r] = lstate[r] * alpha[r] + rs;
    }
#pragma unroll
    for (int dn = 0; dn < 4; dn++)
#pragma unroll
      for (int r = 0; r < 8; r++) o[dn][r] *= alpha[r];

    // --- relayout P (C-layout -> A-layout) through per-wave LDS ---
#pragma unroll
    for (int jn = 0; jn < 4; jn++)
#pragma unroll
      for (int r = 0; r < 8; r++)
        Ps[wave][r + 8 * hl][jn * 16 + nl] = cvt_bf16(s[jn][r]);
    __syncthreads();

    // --- O += P @ V ---
#pragma unroll
    for (int kk = 0; kk < 2; kk++) {
      v16bf pf = load_frag(&Ps[wave][0][kk * 32], 64, lane);
#pragma unroll
      for (int dn = 0; dn < 4; dn++) {
        v16bf vf = load_frag(&Vt[dn * 16][kk * 32], 64, lane);
        o[dn] = wmma_bf16(pf, vf, o[dn]);
      }
    }
  }

  // --- epilogue: y[b, t, h*64 + d] = O / l ---
  const int b = bh >> 4;
  const int h = bh & 15;
#pragma unroll
  for (int r = 0; r < 8; r++) {
    float inv = 1.0f / lstate[r];
    int gq = qw + r + 8 * hl;
    float* yrow = y + ((size_t)(b * T_ + gq)) * D_ + h * HD_;
#pragma unroll
    for (int dn = 0; dn < 4; dn++) yrow[dn * 16 + nl] = o[dn][r] * inv;
  }
}

// ---------------------------------------------------------------------------
extern "C" void kernel_launch(void* const* d_in, const int* in_sizes, int n_in,
                              void* d_out, int out_size, void* d_ws, size_t ws_size,
                              hipStream_t stream) {
  const float* x      = (const float*)d_in[0];
  const float* w_attn = (const float*)d_in[1];
  const float* b_attn = (const float*)d_in[2];
  const float* w_proj = (const float*)d_in[3];
  const float* b_proj = (const float*)d_in[4];
  float* out = (float*)d_out;

  const size_t qkv_elems = (size_t)B_ * H_ * T_ * HD_;   // 4,194,304
  const size_t qkv_bytes = qkv_elems * sizeof(bf16);     // 8 MB each
  char* ws = (char*)d_ws;
  bf16* q_ws = (bf16*)(ws);
  bf16* k_ws = (bf16*)(ws + qkv_bytes);
  bf16* v_ws = (bf16*)(ws + 2 * qkv_bytes);
  float* y_ws = (float*)(ws + 3 * qkv_bytes);            // 16 MB

  qkv_gemm_kernel<<<dim3(N3_ / 128, M_ / 128), 256, 0, stream>>>(
      x, w_attn, b_attn, q_ws, k_ws, v_ws);

  attn_kernel<<<dim3(T_ / 128, B_ * H_), 256, 0, stream>>>(
      q_ws, k_ws, v_ws, y_ws);

  proj_gemm_kernel<<<dim3(D_ / 128, M_ / 128), 256, 0, stream>>>(
      y_ws, w_proj, b_proj, out);
}
